// SemiLSTM_47442208751619
// MI455X (gfx1250) — matrix operand order repaired and checked by reading, not compiled, not run on hardware
//
#include <hip/hip_runtime.h>

// ---------------------------------------------------------------------------
// LSTM forward for MI455X (gfx1250, wave32, WMMA).
//   B=32, T=512, I=H=1024, gates G=4H=4096.
// Phase 1: x_proj = xs @ W_ih^T + (b_ih+b_hh), stored transposed [T, 4H, B]
//          so both the GEMM epilogue store and the recurrence load are b128.
// Phase 2: persistent kernel, 512 sequential steps with grid barrier:
//          gates = x_proj[t] + h @ W_hh^T  (bf16 WMMA, f32 accum).
// Both GEMM K-loops are software-pipelined (double-buffered fragments) so the
// scheduler can overlap global loads with v_wmma instead of s_wait_loadcnt 0.
// ---------------------------------------------------------------------------

constexpr int BB = 32;      // batch
constexpr int TT = 512;     // timesteps
constexpr int II = 1024;    // input features
constexpr int HH = 1024;    // hidden size
constexpr int GG = 4 * HH;  // gate rows (i,f,g,o)

constexpr int REC_BLOCKS = 32;  // persistent recurrence blocks (128 waves total)

typedef __bf16 bf16;
typedef __attribute__((ext_vector_type(16))) __bf16 v16bf;
typedef __attribute__((ext_vector_type(8)))  __bf16 v8bf;
typedef __attribute__((ext_vector_type(8)))  float  v8f;

union FragU { v16bf v; v8bf h[2]; };

// ---- WMMA wrapper: D = A(16x32 bf16) * B(32x16 bf16) + C(16x16 f32) -------
__device__ __forceinline__ v8f wmma_bf16(v16bf a, v16bf b, v8f c) {
#if defined(__HIP_DEVICE_COMPILE__)
  return __builtin_amdgcn_wmma_f32_16x16x32_bf16(
      /*neg_a=*/false, a, /*neg_b=*/false, b,
      /*c_mod=*/(short)0, c, /*reuse_a=*/false, /*reuse_b=*/false);
#else
  (void)a; (void)b; return c;
#endif
}

// A-fragment, 16x32 bf16 row-major.  ISA layout: lanes 0-15 hold
// K {k0..k0+7, k0+16..k0+23}; lanes 16-31 hold K {k0+8..k0+15, k0+24..k0+31}.
__device__ __forceinline__ v16bf load_frag_A(const bf16* rowbase, int k0, int sel) {
  FragU f;
  const bf16* p = rowbase + k0 + sel * 8;
  f.h[0] = *(const v8bf*)(p);
  f.h[1] = *(const v8bf*)(p + 16);
  return f.v;
}

// B-fragment, 32x16 bf16 where B[k][n] = W[n][k] (W row-major [N][K]).
// Lane owns column n; lanes 0-15: K {k0..k0+7, k0+16..k0+23}, lanes 16-31: K {k0+8..k0+15, k0+24..k0+31}.
__device__ __forceinline__ v16bf load_frag_B(const bf16* rowbase, int k0, int sel) {
  FragU f;
  const bf16* p = rowbase + k0 + sel * 8;
  f.h[0] = *(const v8bf*)(p);
  f.h[1] = *(const v8bf*)(p + 16);
  return f.v;
}

// Software-pipelined 16x16 (x4 gate columns) GEMM over K: fragments for
// iteration k+1 are loaded while iteration k's 4 WMMAs execute; last
// iteration peeled.  acc[4] accumulate in f32.
template <int K>
__device__ __forceinline__ void gemm4_pipelined(const bf16* arow,
                                                const bf16* b0, const bf16* b1,
                                                const bf16* b2, const bf16* b3,
                                                int sel, v8f acc[4]) {
  v16bf a  = load_frag_A(arow, 0, sel);
  v16bf w0 = load_frag_B(b0, 0, sel);
  v16bf w1 = load_frag_B(b1, 0, sel);
  v16bf w2 = load_frag_B(b2, 0, sel);
  v16bf w3 = load_frag_B(b3, 0, sel);
  for (int k0 = 32; k0 < K; k0 += 32) {
    v16bf an  = load_frag_A(arow, k0, sel);
    v16bf w0n = load_frag_B(b0, k0, sel);
    v16bf w1n = load_frag_B(b1, k0, sel);
    v16bf w2n = load_frag_B(b2, k0, sel);
    v16bf w3n = load_frag_B(b3, k0, sel);
    acc[0] = wmma_bf16(a, w0, acc[0]);
    acc[1] = wmma_bf16(a, w1, acc[1]);
    acc[2] = wmma_bf16(a, w2, acc[2]);
    acc[3] = wmma_bf16(a, w3, acc[3]);
    a = an; w0 = w0n; w1 = w1n; w2 = w2n; w3 = w3n;
  }
  acc[0] = wmma_bf16(a, w0, acc[0]);
  acc[1] = wmma_bf16(a, w1, acc[1]);
  acc[2] = wmma_bf16(a, w2, acc[2]);
  acc[3] = wmma_bf16(a, w3, acc[3]);
}

__device__ __forceinline__ float sigf(float x)   { return 1.0f / (1.0f + __expf(-x)); }
__device__ __forceinline__ float tanhf_f(float x){ return 2.0f / (1.0f + __expf(-2.0f * x)) - 1.0f; }

// ---- Global spin barrier across the persistent grid ------------------------
__device__ __forceinline__ void grid_barrier(int* cnt, int* gen, int nblk) {
  __threadfence();
  __syncthreads();
  if (threadIdx.x == 0) {
    volatile int* vgen = (volatile int*)gen;
    int g = *vgen;
    if (atomicAdd(cnt, 1) == nblk - 1) {
      *(volatile int*)cnt = 0;
      __threadfence();
      atomicAdd(gen, 1);
    } else {
      while (*vgen == g) {
#if defined(__HIP_DEVICE_COMPILE__)
        __builtin_amdgcn_s_sleep(2);
#endif
      }
    }
  }
  __syncthreads();
}

// ---- Prep kernels ----------------------------------------------------------
__global__ void __launch_bounds__(256)
cvt_f32_bf16(const float* __restrict__ src, bf16* __restrict__ dst, int n) {
  int i = blockIdx.x * blockDim.x + threadIdx.x;
  int stride = gridDim.x * blockDim.x;
  for (; i < n; i += stride) dst[i] = (bf16)src[i];
}

__global__ void __launch_bounds__(256)
bias_init(const float* __restrict__ bih, const float* __restrict__ bhh,
          float* __restrict__ b) {
  int i = blockIdx.x * blockDim.x + threadIdx.x;
  if (i < GG) b[i] = bih[i] + bhh[i];
}

__global__ void __launch_bounds__(256)
state_init(float* __restrict__ c, bf16* __restrict__ h0, int* __restrict__ bar) {
  int i = blockIdx.x * blockDim.x + threadIdx.x;
  if (i < BB * HH) { c[i] = 0.0f; h0[i] = (bf16)0.0f; }
  if (i == 0) { bar[0] = 0; bar[1] = 0; }
}

// ---- Phase 1: x_proj (transposed [T, 4H, B]) = xs_bf @ W_ih^T + bias -------
// One wave per (16-row m-tile, 64-col n-group): 4 accumulators reuse the A
// fragment 4x.  1024 m-tiles x 64 n-groups = 65536 waves = 8192 blocks.
// A 16-row m-tile spans exactly one timestep t = m0/32 with batch rows
// b0 = m0%32 .. b0+15, so the transposed store is contiguous in batch.
__global__ void __launch_bounds__(256)
xproj_gemm(const bf16* __restrict__ xsb, const bf16* __restrict__ Wih,
           const float* __restrict__ bias, bf16* __restrict__ xproj) {
  const int lane = threadIdx.x & 31;
  const int wave = threadIdx.x >> 5;
  const int wid  = blockIdx.x * 8 + wave;
  const int ng    = wid & 63;
  const int mtile = wid >> 6;
  const int m0 = mtile * 16;
  const int n0 = ng * 64;
  const int l15 = lane & 15;
  const int sel = lane >> 4;

  // A row for this lane: global row m = t*B + b  ->  xs[b, t, :]
  const int m  = m0 + l15;
  const int bb = m & (BB - 1);
  const int tt = m >> 5;
  const bf16* arow = xsb + ((size_t)bb * TT + tt) * II;

  const bf16* brow0 = Wih + (size_t)(n0 + 0 * 16 + l15) * II;
  const bf16* brow1 = Wih + (size_t)(n0 + 1 * 16 + l15) * II;
  const bf16* brow2 = Wih + (size_t)(n0 + 2 * 16 + l15) * II;
  const bf16* brow3 = Wih + (size_t)(n0 + 3 * 16 + l15) * II;

  v8f acc[4] = {};
  gemm4_pipelined<II>(arow, brow0, brow1, brow2, brow3, sel, acc);

  // Store: this lane's C column is col = n0+q*16+l15; its 8 rows are batch
  // rows b0..b0+7 of timestep t -> one contiguous v8bf store per gate.
  const int t  = m0 >> 5;
  const int b0 = (m0 & 31) + sel * 8;
#pragma unroll
  for (int q = 0; q < 4; ++q) {
    const int col = n0 + q * 16 + l15;
    const float bv = bias[col];
    v8bf pk;
#pragma unroll
    for (int r = 0; r < 8; ++r) pk[r] = (bf16)(acc[q][r] + bv);
    *(v8bf*)(xproj + ((size_t)t * GG + col) * BB + b0) = pk;
  }
}

// ---- Phase 2: persistent recurrence ----------------------------------------
// 128 waves: wave owns (bt in 0..1, jt in 0..63) -> 16 batch rows x 16 hidden
// cols, all 4 gates.  W_hh (8 MB bf16) stays L2-resident across all 512 steps.
__global__ void __launch_bounds__(128)
lstm_rec(const bf16* __restrict__ Whh, const bf16* __restrict__ xproj,
         float* __restrict__ cst, bf16* __restrict__ hbuf0, bf16* __restrict__ hbuf1,
         float* __restrict__ rst, float* __restrict__ hfin, float* __restrict__ cfin,
         int* __restrict__ bar) {
  const int lane = threadIdx.x & 31;
  const int wave = threadIdx.x >> 5;
  const int wid  = blockIdx.x * 4 + wave;   // 0..127
  const int bt = wid >> 6;                  // batch tile 0..1
  const int jt = wid & 63;                  // hidden tile 0..63
  const int l15 = lane & 15;
  const int sel = lane >> 4;
  const int j = jt * 16 + l15;              // this lane's hidden column

  const bf16* w0 = Whh + (size_t)(0 * HH + j) * HH;
  const bf16* w1 = Whh + (size_t)(1 * HH + j) * HH;
  const bf16* w2 = Whh + (size_t)(2 * HH + j) * HH;
  const bf16* w3 = Whh + (size_t)(3 * HH + j) * HH;

  const int arow_b = bt * 16 + l15;         // batch row this lane loads for A
  const int crow0  = bt * 16 + sel * 8;     // first batch row this lane stores

  for (int t = 0; t < TT; ++t) {
    const bf16* hprev = (t & 1) ? hbuf1 : hbuf0;
    bf16*       hnext = (t & 1) ? hbuf0 : hbuf1;
    const bf16* arow  = hprev + (size_t)arow_b * HH;

    // Prefetch this step's x_proj slices (one per gate) while the GEMM runs.
#pragma unroll
    for (int q = 0; q < 4; ++q)
      __builtin_prefetch(
          (const void*)(xproj + ((size_t)t * GG + q * HH + j) * BB + crow0), 0, 2);

    v8f acc[4] = {};
    gemm4_pipelined<HH>(arow, w0, w1, w2, w3, sel, acc);

    // x_proj is [T, 4H, B]: this lane's 8 batch rows per gate are contiguous.
    v8bf xq[4];
#pragma unroll
    for (int q = 0; q < 4; ++q)
      xq[q] = *(const v8bf*)(xproj + ((size_t)t * GG + q * HH + j) * BB + crow0);

#pragma unroll
    for (int r = 0; r < 8; ++r) {
      float gi = acc[0][r] + (float)xq[0][r];
      float gf = acc[1][r] + (float)xq[1][r];
      float gg = acc[2][r] + (float)xq[2][r];
      float go = acc[3][r] + (float)xq[3][r];
      const float iv = sigf(gi);
      const float fv = sigf(gf);
      const float gv = tanhf_f(gg);
      const float ov = sigf(go);
      const int brow = crow0 + r;
      const int ci = brow * HH + j;
      const float cnew = fv * cst[ci] + iv * gv;
      cst[ci] = cnew;
      const float hv = ov * tanhf_f(cnew);
      rst[((size_t)brow * TT + t) * HH + j] = hv;
      hnext[ci] = (bf16)hv;
      if (t == TT - 1) { hfin[ci] = hv; cfin[ci] = cnew; }
    }

    grid_barrier(bar, bar + 1, REC_BLOCKS);
  }
}

// ---------------------------------------------------------------------------
extern "C" void kernel_launch(void* const* d_in, const int* in_sizes, int n_in,
                              void* d_out, int out_size, void* d_ws, size_t ws_size,
                              hipStream_t stream) {
  (void)in_sizes; (void)n_in; (void)out_size; (void)ws_size;
  const float* xs  = (const float*)d_in[0];   // [B,T,I]
  const float* Wih = (const float*)d_in[1];   // [4H,I]
  const float* Whh = (const float*)d_in[2];   // [4H,H]
  const float* bih = (const float*)d_in[3];   // [4H]
  const float* bhh = (const float*)d_in[4];   // [4H]
  float* out = (float*)d_out;                 // rst[B,T,H] | h[B,H] | c[B,H]
  float* hfin = out + (size_t)BB * TT * HH;
  float* cfin = hfin + (size_t)BB * HH;

  char* ws = (char*)d_ws;
  size_t off = 0;
  bf16* Wih_bf = (bf16*)(ws + off); off += (size_t)GG * II * sizeof(bf16);      //   8 MiB
  bf16* Whh_bf = (bf16*)(ws + off); off += (size_t)GG * HH * sizeof(bf16);      //   8 MiB
  bf16* xs_bf  = (bf16*)(ws + off); off += (size_t)BB * TT * II * sizeof(bf16); //  32 MiB
  bf16* xproj  = (bf16*)(ws + off); off += (size_t)TT * GG * BB * sizeof(bf16); // 128 MiB
  float* biasc = (float*)(ws + off); off += (size_t)GG * sizeof(float);
  float* cst   = (float*)(ws + off); off += (size_t)BB * HH * sizeof(float);
  bf16* hbuf0  = (bf16*)(ws + off);  off += (size_t)BB * HH * sizeof(bf16);
  bf16* hbuf1  = (bf16*)(ws + off);  off += (size_t)BB * HH * sizeof(bf16);
  int*  bar    = (int*)(ws + off);   off += 256;

  // Precision prep: f32 -> bf16 for all matrix operands, combine biases.
  cvt_f32_bf16<<<2048, 256, 0, stream>>>(Wih, Wih_bf, GG * II);
  cvt_f32_bf16<<<2048, 256, 0, stream>>>(Whh, Whh_bf, GG * HH);
  cvt_f32_bf16<<<4096, 256, 0, stream>>>(xs,  xs_bf,  BB * TT * II);
  bias_init<<<16, 256, 0, stream>>>(bih, bhh, biasc);
  state_init<<<128, 256, 0, stream>>>(cst, hbuf0, bar);

  // Phase 1: 16384x4096x1024 bf16 GEMM (137 GFLOP) on WMMA.
  xproj_gemm<<<8192, 256, 0, stream>>>(xs_bf, Wih_bf, biasc, xproj);

  // Phase 2: persistent recurrence, 512 steps with device-wide barrier.
  lstm_rec<<<REC_BLOCKS, 128, 0, stream>>>(Whh_bf, xproj, cst, hbuf0, hbuf1,
                                           out, hfin, cfin, bar);
}
